// WMSA_40321152975287
// MI455X (gfx1250) — matrix operand order repaired and checked by reading, not compiled, not run on hardware
//
#include <hip/hip_runtime.h>

#define WSZ 8
#define HD 32
#define NH 6
#define CCH 192
#define IMG 256
#define NWIN_SIDE 32
#define NWIN (NWIN_SIDE*NWIN_SIDE)
#define BATCH 4

typedef __attribute__((ext_vector_type(16))) _Float16 v16h;
typedef __attribute__((ext_vector_type(8)))  _Float16 v8h;
typedef __attribute__((ext_vector_type(8)))  float    v8f;
typedef __attribute__((ext_vector_type(4)))  unsigned int u32x4;
typedef __attribute__((ext_vector_type(8)))  unsigned int u32x8;

// ---------------------------------------------------------------------------
// Pack fp32 row-major weight [N][K] into f16 WMMA B-fragments:
// packed[((nt*(K/32)+kt)*32 + lane)*16 + i]
// lane 0-15 : n = nt*16+lane,    k = kt*32 + i        (K 0..15)
// lane 16-31: n = nt*16+lane-16, k = kt*32 + 16 + i   (K 16..31)
// Each lane's 16 halves are contiguous -> single 32B global load in main loop.
// ---------------------------------------------------------------------------
__global__ void pack_w_kernel(const float* __restrict__ w, _Float16* __restrict__ packed,
                              int N, int K) {
    int idx = blockIdx.x * blockDim.x + threadIdx.x;
    if (idx >= N * K) return;
    int KT = K >> 5;
    int i    = idx & 15;
    int lane = (idx >> 4) & 31;
    int rem  = idx >> 9;
    int kt   = rem % KT;
    int nt   = rem / KT;
    int n = nt * 16 + (lane & 15);
    int k = kt * 32 + ((lane >> 4) << 4) + i;
    packed[idx] = (_Float16)w[n * K + k];
}

// A-fragment (16x32 f16) from row-major f16 (stride in elements).
__device__ __forceinline__ v16h load_a_frag(const _Float16* base, int stride,
                                            int row0, int k0, int lane) {
    int r  = row0 + (lane & 15);
    int kk = k0 + ((lane >> 4) << 4);
    const _Float16* p = base + r * stride + kk;
    v8h lo = *(const v8h*)p;
    v8h hi = *(const v8h*)(p + 8);
    v16h o;
#pragma unroll
    for (int i = 0; i < 8; ++i) { o[i] = lo[i]; o[i + 8] = hi[i]; }
    return o;
}

// B-fragment (32x16 f16) from COLUMN-major storage ([n][k], contiguous in k).
__device__ __forceinline__ v16h load_b_frag(const _Float16* base, int stride,
                                            int k0, int n0, int lane) {
    int n  = n0 + (lane & 15);
    int kk = k0 + ((lane >> 4) << 4);
    return *(const v16h*)(base + n * stride + kk);
}

__device__ __forceinline__ v8f wmma_f16(v16h a, v16h b, v8f c) {
    return __builtin_amdgcn_wmma_f32_16x16x32_f16(false, a, false, b, (short)0, c,
                                                  false, false);
}

// ---------------------------------------------------------------------------
// Fused W-MSA: one workgroup (8 wave32) per 8x8 window.
// ---------------------------------------------------------------------------
__launch_bounds__(256)
__global__ void wmsa_fused_kernel(const float* __restrict__ x,
                                  const _Float16* __restrict__ wqkv,   // packed B-frags
                                  const float* __restrict__ bqkv,
                                  const _Float16* __restrict__ wproj,  // packed B-frags
                                  const float* __restrict__ bproj,
                                  const float* __restrict__ relb,
                                  float* __restrict__ out) {
    __shared__ _Float16 xw_s[64 * 200];      // x window f16; reused for attn-out
    __shared__ _Float16 q_s [64 * 200];      // q * scale, [pix][h*32+d]
    __shared__ _Float16 k_s [64 * 200];      // k,         [pix][h*32+d]
    __shared__ _Float16 vT_s[192 * 72];      // v transposed [h*32+d][pix]
    __shared__ _Float16 probs_s[8 * 16 * 72];// per-wave 16x64 probs strip
    __shared__ float    relb_s[NH * 15 * 15];// rel-bias table, filled by TDM

    const int tid   = threadIdx.x;
    const int lane  = tid & 31;
    // Force wave id scalar so per-tile control flow stays on the SALU side.
    const int wave  = __builtin_amdgcn_readfirstlane(tid >> 5);
    const int lhalf = lane >> 4;
    const int l16   = lane & 15;

    const int win = blockIdx.x;
    const int b   = win / NWIN;
    const int wi  = win % NWIN;
    const int wy  = wi / NWIN_SIDE;
    const int wx  = wi % NWIN_SIDE;
    const long base_px = ((long)b * IMG + wy * WSZ) * IMG + wx * WSZ;

    // ---- TDM: DMA the rel-bias table (1350 fp32) into LDS, wave 0 only ----
    if (wave == 0) {
        unsigned long long ga = (unsigned long long)(uintptr_t)relb;
        unsigned lds_off = (unsigned)(uintptr_t)relb_s;
        // D# group0: count=1 | lds_addr | global_addr[56:0] | type=2
        u32x4 g0 = { 1u, lds_off, (unsigned)ga,
                     (unsigned)(ga >> 32) | 0x80000000u };
        // D# group1: data_size=4B; tensor_dim0=tile_dim0=1350; dim1=tile1=1;
        //            tensor_dim0_stride=1350; no pad/iterate/multicast.
        u32x8 g1 = { 0x00020000u,            // mask=0, data_size=2 (4B)
                     0x05460000u,            // tensor_dim0[15:0]=1350 << 16
                     0x00010000u,            // tensor_dim0[31:16]=0, tensor_dim1=1
                     0x05460000u,            // tensor_dim1[31:16]=0, tile_dim0=1350
                     0x00000001u,            // tile_dim1=1, tile_dim2=0
                     0x00000546u,            // tensor_dim0_stride=1350
                     0x00000000u, 0x00000000u };
        asm volatile("tensor_load_to_lds %0, %1\n\t"
                     "s_wait_tensorcnt 0x0"
                     :: "s"(g0), "s"(g1) : "memory");
    }

    // ---- Phase 0: load 64x192 fp32 window -> f16 LDS ----------------------
    for (int i = tid; i < 64 * 48; i += 256) {
        int p  = i / 48;
        int c4 = (i % 48) << 2;
        int py = p >> 3, px = p & 7;
        const float4 v = *(const float4*)(x + (base_px + py * IMG + px) * CCH + c4);
        _Float16* d = &xw_s[p * 200 + c4];
        d[0] = (_Float16)v.x; d[1] = (_Float16)v.y;
        d[2] = (_Float16)v.z; d[3] = (_Float16)v.w;
    }
    __syncthreads();

    // ---- Phase 1: QKV GEMM  (64x192)@(192x576) ----------------------------
    const float scale = 0.17677669529663687f;  // HD^-0.5
    for (int t = wave; t < 144; t += 8) {
        int mt = t & 3;
        int nt = t >> 2;
        v8f acc = {};
#pragma unroll
        for (int kt = 0; kt < 6; ++kt) {
            v16h a = load_a_frag(xw_s, 200, mt * 16, kt * 32, lane);
            const _Float16* bp = wqkv + (((nt * 6 + kt) * 32) + lane) * 16;
            v16h bf = *(const v16h*)bp;
            acc = wmma_f16(a, bf, acc);
        }
        if (t + 8 < 144)
            __builtin_prefetch(wqkv + ((((t + 8) >> 2) * 6 * 32) + lane) * 16, 0, 1);

        int n = nt * 16 + l16;
        float bias = bqkv[n];
        if (nt < 12) {                       // q (scaled)
#pragma unroll
            for (int r = 0; r < 8; ++r) {
                int row = mt * 16 + r + (lhalf << 3);
                q_s[row * 200 + n] = (_Float16)((acc[r] + bias) * scale);
            }
        } else if (nt < 24) {                // k
            int nn = n - 192;
#pragma unroll
            for (int r = 0; r < 8; ++r) {
                int row = mt * 16 + r + (lhalf << 3);
                k_s[row * 200 + nn] = (_Float16)(acc[r] + bias);
            }
        } else {                             // v, stored transposed
            int nn = n - 384;
#pragma unroll
            for (int r = 0; r < 8; ++r) {
                int row = mt * 16 + r + (lhalf << 3);
                vT_s[nn * 72 + row] = (_Float16)(acc[r] + bias);
            }
        }
    }
    __syncthreads();

    // ---- Phase 2: attention, 24 strips (6 heads x 4 row-tiles) ------------
    for (int s = wave; s < 24; s += 8) {
        int h  = s >> 2;
        int m0 = (s & 3) << 4;

        // sim strip 16x64: one K=32 WMMA per 16x16 tile
        v16h aq = load_a_frag(q_s + h * HD, 200, m0, 0, lane);
        v8f acc[4];
#pragma unroll
        for (int nt2 = 0; nt2 < 4; ++nt2) {
            v16h bk = load_b_frag(k_s + h * HD, 200, 0, nt2 * 16, lane);
            v8f z = {};
            acc[nt2] = wmma_f16(aq, bk, z);
        }

        // relative position bias (from LDS, TDM-loaded)
        int p_base = m0 + (lhalf << 3);
#pragma unroll
        for (int nt2 = 0; nt2 < 4; ++nt2) {
            int qq = nt2 * 16 + l16;
#pragma unroll
            for (int r = 0; r < 8; ++r) {
                int p  = p_base + r;
                int r0 = (p >> 3) - (qq >> 3) + 7;
                int r1 = (p & 7) - (qq & 7) + 7;
                acc[nt2][r] += relb_s[h * 225 + r0 * 15 + r1];
            }
        }

        // in-register softmax: rows live in 16-lane halves
        _Float16* pw = &probs_s[wave * 16 * 72];
#pragma unroll
        for (int r = 0; r < 8; ++r) {
            float m = fmaxf(fmaxf(acc[0][r], acc[1][r]), fmaxf(acc[2][r], acc[3][r]));
            m = fmaxf(m, __shfl_xor(m, 1, 32));
            m = fmaxf(m, __shfl_xor(m, 2, 32));
            m = fmaxf(m, __shfl_xor(m, 4, 32));
            m = fmaxf(m, __shfl_xor(m, 8, 32));
            float sum = 0.f;
#pragma unroll
            for (int nt2 = 0; nt2 < 4; ++nt2) {
                acc[nt2][r] = __expf(acc[nt2][r] - m);
                sum += acc[nt2][r];
            }
            sum += __shfl_xor(sum, 1, 32);
            sum += __shfl_xor(sum, 2, 32);
            sum += __shfl_xor(sum, 4, 32);
            sum += __shfl_xor(sum, 8, 32);
            float inv = __builtin_amdgcn_rcpf(sum);
            int lr = r + (lhalf << 3);
#pragma unroll
            for (int nt2 = 0; nt2 < 4; ++nt2)
                pw[lr * 72 + nt2 * 16 + l16] = (_Float16)(acc[nt2][r] * inv);
        }

        // wave-private LDS write->read: drain DS counter explicitly
        asm volatile("s_wait_dscnt 0x0" ::: "memory");

        // PV: (16x64 probs) @ (64x32 v)  -> two 16x16 tiles, K split in 2
        v8f o0 = {}, o1 = {};
#pragma unroll
        for (int kh = 0; kh < 2; ++kh) {
            v16h ap = load_a_frag(pw, 72, 0, kh * 32, lane);
            v16h b0 = load_b_frag(vT_s + (h * HD) * 72, 72, kh * 32, 0,  lane);
            v16h b1 = load_b_frag(vT_s + (h * HD) * 72, 72, kh * 32, 16, lane);
            o0 = wmma_f16(ap, b0, o0);
            o1 = wmma_f16(ap, b1, o1);
        }
#pragma unroll
        for (int r = 0; r < 8; ++r) {
            int row = m0 + r + (lhalf << 3);
            xw_s[row * 200 + h * HD + l16]      = (_Float16)o0[r];
            xw_s[row * 200 + h * HD + 16 + l16] = (_Float16)o1[r];
        }
    }
    __syncthreads();

    // ---- Phase 3: output projection (64x192)@(192x192) + bias -> global --
    for (int t = wave; t < 48; t += 8) {
        int nt = t % 12;
        int mt = t / 12;
        v8f acc = {};
#pragma unroll
        for (int kt = 0; kt < 6; ++kt) {
            v16h a = load_a_frag(xw_s, 200, mt * 16, kt * 32, lane);
            v16h bf = *(const v16h*)(wproj + (((nt * 6 + kt) * 32) + lane) * 16);
            acc = wmma_f16(a, bf, acc);
        }
        int c = nt * 16 + l16;
        float bias = bproj[c];
#pragma unroll
        for (int r = 0; r < 8; ++r) {
            int p  = mt * 16 + r + (lhalf << 3);
            int py = p >> 3, px = p & 7;
            out[(base_px + py * IMG + px) * CCH + c] = acc[r] + bias;
        }
    }
}

extern "C" void kernel_launch(void* const* d_in, const int* in_sizes, int n_in,
                              void* d_out, int out_size, void* d_ws, size_t ws_size,
                              hipStream_t stream) {
    (void)in_sizes; (void)n_in; (void)out_size; (void)ws_size;
    const float* x      = (const float*)d_in[0];
    const float* qkv_w  = (const float*)d_in[1];
    const float* qkv_b  = (const float*)d_in[2];
    const float* proj_w = (const float*)d_in[3];
    const float* proj_b = (const float*)d_in[4];
    const float* rel_b  = (const float*)d_in[5];

    _Float16* wqkv_p  = (_Float16*)d_ws;               // 576*192 halves
    _Float16* wproj_p = wqkv_p + 576 * 192;            // 192*192 halves

    pack_w_kernel<<<(576 * 192 + 255) / 256, 256, 0, stream>>>(qkv_w, wqkv_p, 576, 192);
    pack_w_kernel<<<(192 * 192 + 255) / 256, 256, 0, stream>>>(proj_w, wproj_p, 192, 192);

    wmsa_fused_kernel<<<BATCH * NWIN, 256, 0, stream>>>(
        x, wqkv_p, qkv_b, wproj_p, proj_b, rel_b, (float*)d_out);
}
